// RWKV5r5_AttentionSubLayer_88854283420281
// MI455X (gfx1250) — matrix operand order repaired and checked by reading, not compiled, not run on hardware
//
#include <hip/hip_runtime.h>
#include <hip/hip_bf16.h>

// RWKV5 attention sublayer for MI455X (gfx1250, wave32, WMMA, async LDS loads).
// cast weights f32->f16 -> token-shift mix -> 4x double-buffered WMMA GEMMs
// (r,k,v,g+silu) -> i-split recurrent scan (raw out) -> wave-parallel LN+gate
// -> final WMMA GEMM (f32 out).

#define B_ 8
#define T_ 2048
#define C_ 1024
#define H_ 16
#define S_ 64

typedef __attribute__((ext_vector_type(16))) _Float16     v16h;
typedef __attribute__((ext_vector_type(8)))  float        v8f;
typedef __attribute__((ext_vector_type(8)))  unsigned int v8u;

union FragCvt { v8u u; v16h h; };

// Async global->LDS path (CDNA5 GLOBAL_LOAD_ASYNC_TO_LDS_B128, ASYNCcnt).
// ROCm 7.2 clang-22 signature (from probe diagnostic):
//   void __builtin_amdgcn_global_load_async_to_lds_b128(
//       int4 addrspace(1)* src, int4 addrspace(3)* dst, imm offset, imm cpol)
#if defined(__HIP_DEVICE_COMPILE__) && defined(__gfx1250__) && \
    __has_builtin(__builtin_amdgcn_global_load_async_to_lds_b128) && \
    __has_builtin(__builtin_amdgcn_s_wait_asynccnt)
#define ASYNC_LDS 1
typedef int v4i __attribute__((vector_size(16)));
typedef __attribute__((address_space(1))) v4i gv4i;
typedef __attribute__((address_space(3))) v4i lv4i;
#else
#define ASYNC_LDS 0
#endif

// ---------------- weight cast f32 -> f16 ----------------
__global__ void cast_f32_f16_kernel(const float* __restrict__ in,
                                    _Float16* __restrict__ out, int n) {
  int i = blockIdx.x * blockDim.x + threadIdx.x;
  if (i < n) out[i] = (_Float16)in[i];
}

// ---------------- token-shift mix ----------------
__global__ __launch_bounds__(256) void mix_kernel(
    const float* __restrict__ xq,
    const float* __restrict__ m_r, const float* __restrict__ m_k,
    const float* __restrict__ m_v, const float* __restrict__ m_g,
    _Float16* __restrict__ xr, _Float16* __restrict__ xk,
    _Float16* __restrict__ xv, _Float16* __restrict__ xg) {
  size_t idx = (size_t)blockIdx.x * 256 + threadIdx.x;
  int c = (int)(idx & (C_ - 1));
  size_t bt = idx >> 10;                 // / C_
  int t = (int)(bt & (T_ - 1));          // % T_
  float x  = xq[idx];
  float xs = (t > 0) ? xq[idx - C_] : 0.0f;
  float mr = m_r[c], mk = m_k[c], mv = m_v[c], mg = m_g[c];
  xr[idx] = (_Float16)(x * mr + xs * (1.0f - mr));
  xk[idx] = (_Float16)(x * mk + xs * (1.0f - mk));
  xv[idx] = (_Float16)(x * mv + xs * (1.0f - mv));
  xg[idx] = (_Float16)(x * mg + xs * (1.0f - mg));
}

// ---------------- tiled WMMA GEMM: Out = A(MxK) @ W(NxK)^T ----------------
// Double-buffered LDS; async global->LDS when available.
// MODE 0: store f16; MODE 1: store silu(x) f16; MODE 2: store f32.
template <int MODE>
__global__ __launch_bounds__(256, 2) void gemm_wmma_kernel(
    const _Float16* __restrict__ A,   // M x K row-major, f16
    const _Float16* __restrict__ W,   // N x K row-major, f16
    void* __restrict__ OutV,
    int Ntot, int Ktot) {
  __shared__ _Float16 As[2][128][40];   // double-buffered M x K tile
  __shared__ _Float16 Bs[2][128][40];   // double-buffered N x K tile

  const int tid  = threadIdx.x;
  const int lane = tid & 31;
  const int wave = tid >> 5;          // 0..7
  const int wm   = wave >> 1;         // 0..3 : 32-row strip
  const int wn   = wave & 1;          // 0..1 : 64-col strip
  const int half = lane >> 4;         // 0..1
  const int l16  = lane & 15;

  const int bm = blockIdx.x * 128;
  const int bn = blockIdx.y * 128;

  const int ldr = tid >> 1;           // 0..127 row of the tile
  const int seg = (tid & 1) * 16;     // 0 or 16 (halves)

  const _Float16* gA = A + (size_t)(bm + ldr) * Ktot + seg;
  const _Float16* gW = W + (size_t)(bn + ldr) * Ktot + seg;

  // Issue one 128x32 A-tile + B-tile into LDS buffer `buf` (4 x 16B per thread).
  auto issue_tile = [&](int kbase, int buf) {
#if ASYNC_LDS
    __builtin_amdgcn_global_load_async_to_lds_b128(
        (gv4i*)(gA + kbase), (lv4i*)&As[buf][ldr][seg], 0, 0);
    __builtin_amdgcn_global_load_async_to_lds_b128(
        (gv4i*)(gA + kbase + 8), (lv4i*)&As[buf][ldr][seg + 8], 0, 0);
    __builtin_amdgcn_global_load_async_to_lds_b128(
        (gv4i*)(gW + kbase), (lv4i*)&Bs[buf][ldr][seg], 0, 0);
    __builtin_amdgcn_global_load_async_to_lds_b128(
        (gv4i*)(gW + kbase + 8), (lv4i*)&Bs[buf][ldr][seg + 8], 0, 0);
#else
    const float4* sa = (const float4*)(gA + kbase);
    float4 a0 = sa[0], a1 = sa[1];
    const float4* sb = (const float4*)(gW + kbase);
    float4 b0 = sb[0], b1 = sb[1];
    *(float4*)&As[buf][ldr][seg]     = a0;
    *(float4*)&As[buf][ldr][seg + 8] = a1;
    *(float4*)&Bs[buf][ldr][seg]     = b0;
    *(float4*)&Bs[buf][ldr][seg + 8] = b1;
#endif
  };

  v8f acc[2][4];
#pragma unroll
  for (int mi = 0; mi < 2; ++mi)
#pragma unroll
    for (int ni = 0; ni < 4; ++ni) acc[mi][ni] = (v8f){};

  const int nk = Ktot >> 5;  // K / 32
  issue_tile(0, 0);

  for (int kb = 0; kb < nk; ++kb) {
    const int buf = kb & 1;
    if (kb + 1 < nk) issue_tile((kb + 1) << 5, buf ^ 1);
#if ASYNC_LDS
    if (kb + 1 < nk) __builtin_amdgcn_s_wait_asynccnt(4);  // current tile done
    else             __builtin_amdgcn_s_wait_asynccnt(0);
#endif
    __syncthreads();

    // ---- fragment gathers (ISA 16-bit A 16x32 / B 32x16 lane layouts) ----
    v16h afr[2], bfr[4];
#pragma unroll
    for (int mi = 0; mi < 2; ++mi) {
      FragCvt f;
      int m = wm * 32 + mi * 16 + l16;
#pragma unroll
      for (int j = 0; j < 8; ++j) {
        int k = ((j & 4) ? 16 : 0) + half * 8 + (j & 3) * 2;
        f.u[j] = *(const unsigned int*)&As[buf][m][k];
      }
      afr[mi] = f.h;
    }
#pragma unroll
    for (int ni = 0; ni < 4; ++ni) {
      FragCvt f;
      int n = wn * 64 + ni * 16 + l16;
#pragma unroll
      for (int j = 0; j < 8; ++j) {
        int k = ((j & 4) ? 16 : 0) + half * 8 + (j & 3) * 2;
        f.u[j] = *(const unsigned int*)&Bs[buf][n][k];
      }
      bfr[ni] = f.h;
    }

#pragma unroll
    for (int mi = 0; mi < 2; ++mi)
#pragma unroll
      for (int ni = 0; ni < 4; ++ni)
        acc[mi][ni] = __builtin_amdgcn_wmma_f32_16x16x32_f16(
            false, afr[mi], false, bfr[ni], (short)0, acc[mi][ni], false, false);

    __syncthreads();  // all waves done reading buf before it is refilled
  }

  // ---- epilogue: C/D f32 16x16: (vgpr v, lane l) -> row v+8*half, col l16 ----
#pragma unroll
  for (int mi = 0; mi < 2; ++mi) {
#pragma unroll
    for (int ni = 0; ni < 4; ++ni) {
#pragma unroll
      for (int v = 0; v < 8; ++v) {
        int grow = bm + wm * 32 + mi * 16 + half * 8 + v;
        int gcol = bn + wn * 64 + ni * 16 + l16;
        float val = acc[mi][ni][v];
        if (MODE == 1) val = val / (1.0f + __expf(-val));  // silu
        if (MODE == 2)
          ((float*)OutV)[(size_t)grow * Ntot + gcol] = val;
        else
          ((_Float16*)OutV)[(size_t)grow * Ntot + gcol] = (_Float16)val;
      }
    }
  }
}

// ---------------- RWKV5 recurrent scan (raw out, no LN) ----------------
// One block per (b,h), 256 threads: thread (q,j) owns state rows i=16q..16q+15
// of column j. out_j = sum_q [ sum_i r_i s_ij + (sum_i r_i u_i k_i) * v_j ].
__global__ __launch_bounds__(256) void rwkv_scan_kernel(
    const _Float16* __restrict__ r, const _Float16* __restrict__ k,
    const _Float16* __restrict__ v,
    const float* __restrict__ time_decay, const float* __restrict__ time_faaaa,
    _Float16* __restrict__ out) {
  const int bh = blockIdx.x;
  const int b  = bh >> 4;          // / H_
  const int h  = bh & (H_ - 1);
  const int tid = threadIdx.x;
  const int j  = tid & 63;         // value/output column
  const int q  = tid >> 6;         // 0..3 : 16-row state slice

  __shared__ float kt[S_], rt[S_], vt[S_];
  __shared__ float part[S_][4];    // [j][q], 16B rows

  float wv[16], uu[16], s[16];
#pragma unroll
  for (int ii = 0; ii < 16; ++ii) {
    const int i = q * 16 + ii;
    wv[ii] = __expf(-__expf(time_decay[h * S_ + i]));
    uu[ii] = time_faaaa[h * S_ + i];
    s[ii]  = 0.0f;
  }

  for (int t = 0; t < T_; ++t) {
    const size_t idx = ((size_t)(b * T_ + t)) * C_ + (h << 6);
    if (q == 0)      kt[j] = (float)k[idx + j];
    else if (q == 1) rt[j] = (float)r[idx + j];
    else if (q == 2) vt[j] = (float)v[idx + j];
    __syncthreads();

    const float vj = vt[j];
    float po = 0.0f, pa = 0.0f;
#pragma unroll
    for (int ii = 0; ii < 16; ++ii) {
      const float ki = kt[q * 16 + ii];
      const float ri = rt[q * 16 + ii];
      po += ri * s[ii];
      pa += ri * uu[ii] * ki;
      s[ii] = wv[ii] * s[ii] + ki * vj;   // s' = k (x) v + w * s
    }
    part[j][q] = po + pa * vj;
    __syncthreads();

    if (q == 0) {
      float4 p4 = *(const float4*)&part[j][0];
      out[idx + j] = (_Float16)(p4.x + p4.y + p4.z + p4.w);
    }
    __syncthreads();
  }
}

// ---------------- per-head LayerNorm + gate (in place) ----------------
// One wave (32 lanes) per (b,t,h) vector of 64; 2 elements per lane;
// wave32 __shfl_xor reductions.
__global__ __launch_bounds__(256) void ln_gate_kernel(
    _Float16* __restrict__ o,            // raw scan out, overwritten with y*g
    const _Float16* __restrict__ g,      // silu(g)
    const float* __restrict__ ln_w, const float* __restrict__ ln_b) {
  const int tid  = threadIdx.x;
  const int wid  = tid >> 5;
  const int lane = tid & 31;
  const int gh   = blockIdx.x * 8 + wid;       // (b*T + t)*H + h
  const int h    = gh & (H_ - 1);
  const size_t base = (size_t)gh * S_;
  const int e0 = lane * 2;

  float x0 = (float)o[base + e0]     * 0.125f;   // / HEAD_SIZE_DIVISOR
  float x1 = (float)o[base + e0 + 1] * 0.125f;

  float sum = x0 + x1;
#pragma unroll
  for (int m = 1; m < 32; m <<= 1) sum += __shfl_xor(sum, m, 32);
  const float mean = sum * (1.0f / S_);

  const float d0 = x0 - mean, d1 = x1 - mean;
  float var = d0 * d0 + d1 * d1;
#pragma unroll
  for (int m = 1; m < 32; m <<= 1) var += __shfl_xor(var, m, 32);
  const float inv = rsqrtf(var * (1.0f / S_) + 1e-5f);

  const int c0 = (h << 6) + e0;
  float y0 = d0 * inv * ln_w[c0]     + ln_b[c0];
  float y1 = d1 * inv * ln_w[c0 + 1] + ln_b[c0 + 1];
  y0 *= (float)g[base + e0];
  y1 *= (float)g[base + e0 + 1];
  o[base + e0]     = (_Float16)y0;
  o[base + e0 + 1] = (_Float16)y1;
}

extern "C" void kernel_launch(void* const* d_in, const int* in_sizes, int n_in,
                              void* d_out, int out_size, void* d_ws, size_t ws_size,
                              hipStream_t stream) {
  (void)in_sizes; (void)n_in; (void)out_size; (void)ws_size;

  const float* xq   = (const float*)d_in[0];
  // d_in[1] (xk), d_in[2] (xv) are unused by the reference (x = xq).
  const float* tmk  = (const float*)d_in[3];
  const float* tmv  = (const float*)d_in[4];
  const float* tmr  = (const float*)d_in[5];
  const float* tmg  = (const float*)d_in[6];
  const float* tdec = (const float*)d_in[7];
  const float* tfaa = (const float*)d_in[8];
  const float* Wr   = (const float*)d_in[9];
  const float* Wk   = (const float*)d_in[10];
  const float* Wv   = (const float*)d_in[11];
  const float* Wg   = (const float*)d_in[12];
  const float* Wo   = (const float*)d_in[13];
  const float* lnw  = (const float*)d_in[14];
  const float* lnb  = (const float*)d_in[15];

  const size_t WSZ = (size_t)C_ * C_;            // 1 Mi halves per weight
  const size_t ASZ = (size_t)B_ * T_ * C_;       // 16 Mi halves per activation

  _Float16* Wr_h = (_Float16*)d_ws;
  _Float16* Wk_h = Wr_h + WSZ;
  _Float16* Wv_h = Wk_h + WSZ;
  _Float16* Wg_h = Wv_h + WSZ;
  _Float16* Wo_h = Wg_h + WSZ;
  _Float16* bufA = Wo_h + WSZ;                   // 5 activation buffers, aliased
  _Float16* bufB = bufA + ASZ;
  _Float16* bufC = bufB + ASZ;
  _Float16* bufD = bufC + ASZ;
  _Float16* bufE = bufD + ASZ;

  const int M = B_ * T_;                         // 16384

  // 1) weights f32 -> f16
  cast_f32_f16_kernel<<<(int)(WSZ / 256), 256, 0, stream>>>(Wr, Wr_h, (int)WSZ);
  cast_f32_f16_kernel<<<(int)(WSZ / 256), 256, 0, stream>>>(Wk, Wk_h, (int)WSZ);
  cast_f32_f16_kernel<<<(int)(WSZ / 256), 256, 0, stream>>>(Wv, Wv_h, (int)WSZ);
  cast_f32_f16_kernel<<<(int)(WSZ / 256), 256, 0, stream>>>(Wg, Wg_h, (int)WSZ);
  cast_f32_f16_kernel<<<(int)(WSZ / 256), 256, 0, stream>>>(Wo, Wo_h, (int)WSZ);

  // 2) token-shift mix -> xr(bufA), xk(bufB), xv(bufC), xg(bufD)
  mix_kernel<<<(int)(ASZ / 256), 256, 0, stream>>>(xq, tmr, tmk, tmv, tmg,
                                                   bufA, bufB, bufC, bufD);

  // 3) projection GEMMs (WMMA f16->f32, async double-buffered)
  dim3 gg(M / 128, C_ / 128);
  gemm_wmma_kernel<0><<<gg, 256, 0, stream>>>(bufA, Wr_h, bufE, C_, C_);  // r
  gemm_wmma_kernel<0><<<gg, 256, 0, stream>>>(bufB, Wk_h, bufA, C_, C_);  // k
  gemm_wmma_kernel<0><<<gg, 256, 0, stream>>>(bufC, Wv_h, bufB, C_, C_);  // v
  gemm_wmma_kernel<1><<<gg, 256, 0, stream>>>(bufD, Wg_h, bufC, C_, C_);  // g = silu

  // 4) recurrent scan -> raw out (bufD)
  rwkv_scan_kernel<<<B_ * H_, 256, 0, stream>>>(bufE, bufA, bufB,
                                                tdec, tfaa, bufD);

  // 5) LN + gate in place on bufD (reads g from bufC)
  ln_gate_kernel<<<(B_ * T_ * H_) / 8, 256, 0, stream>>>(bufD, bufC, lnw, lnb);

  // 6) output projection -> d_out (f32)
  gemm_wmma_kernel<2><<<gg, 256, 0, stream>>>(bufD, Wo_h, d_out, C_, C_);
}
